// ImageGeneCrossTransformerRoPE_62912680952098
// MI455X (gfx1250) — compile-verified, matrix operand order checked by reading, not run on hardware
//
#include <hip/hip_runtime.h>
#include <hip/hip_bf16.h>
#include <math.h>

// ---------------------------------------------------------------------------
// Problem constants
// ---------------------------------------------------------------------------
#define BB   16
#define NN   1024
#define GG   2048
#define DD   512
#define DFF  1024
#define HH   8
#define HDIM 64
#define CIN  1024
#define MI   (BB * NN)   // 16384 image rows
#define MG   (BB * GG)   // 32768 gene rows
#define LN_EPS 1e-5f
#define ROPE_LN1E4_OVER_HD 0.14391156831212787f   // ln(10000)/64

typedef __attribute__((ext_vector_type(8)))  _Float16 v8h;
typedef __attribute__((ext_vector_type(16))) _Float16 v16h;
typedef __attribute__((ext_vector_type(8)))  float    v8f;

// ---------------------------------------------------------------------------
// WMMA helpers (CDNA5 wave32, V_WMMA_F32_16X16X32_F16)
// ---------------------------------------------------------------------------
__device__ __forceinline__ v8f wmma_f16(v16h a, v16h b, v8f c) {
  // (neg_a, A, neg_b, B, c_mod, C, reuse_a, reuse_b)
  return __builtin_amdgcn_wmma_f32_16x16x32_f16(false, a, false, b, (short)0, c,
                                                false, false);
}

// Load a 16x32 f16 fragment (A layout; B layout is identical when the matrix
// is stored transposed, i.e. [cols][K] row-major).
// ISA layout: lanes 0-15 hold row=lane, K halves {k0..k0+7, k0+16..k0+23};
// lanes 16-31 hold row=lane-16, K halves {k0+8..k0+15, k0+24..k0+31}.
__device__ __forceinline__ v16h load_frag(const _Float16* tile, int ld, int k0) {
  const int lane = threadIdx.x & 31;
  const int lr = lane & 15;
  const int hf = lane >> 4;
  const _Float16* p = tile + (size_t)lr * ld + k0 + hf * 8;
  v8h lo = *(const v8h*)p;
  v8h hi = *(const v8h*)(p + 16);
  v16h r;
#pragma unroll
  for (int i = 0; i < 8; ++i) { r[i] = lo[i]; r[8 + i] = hi[i]; }
  return r;
}

// ---------------------------------------------------------------------------
// Small conversion kernels
// ---------------------------------------------------------------------------
__global__ void cvt_f32_to_f16(const float* __restrict__ s,
                               _Float16* __restrict__ d, int n) {
  int i = blockIdx.x * 256 + threadIdx.x;
  if (i < n) d[i] = (_Float16)s[i];
}

// src [K][N] f32 (row-major)  ->  dst [N][K] f16
__global__ void cvt_transpose_w(const float* __restrict__ s,
                                _Float16* __restrict__ d, int K, int N) {
  int i = blockIdx.x * 256 + threadIdx.x;
  if (i >= K * N) return;
  int k = i / N, n = i - k * N;
  d[(size_t)n * K + k] = (_Float16)s[i];
}

// ---------------------------------------------------------------------------
// Gene embedding + layernorm (elementwise, one row per block)
// ---------------------------------------------------------------------------
__global__ void gene_embed_ln(const float* __restrict__ gv,
                              const float* __restrict__ gw,
                              const float* __restrict__ gb,
                              const float* __restrict__ gamma,
                              const float* __restrict__ beta,
                              float* __restrict__ out32,
                              _Float16* __restrict__ out16) {
  __shared__ float red[18];
  const int row = blockIdx.x;
  const float x = gv[row];
  float vals[2], sum = 0.f, sq = 0.f;
#pragma unroll
  for (int i = 0; i < 2; ++i) {
    int c = threadIdx.x + i * 256;
    float v = x * gw[c] + gb[c];
    vals[i] = v; sum += v; sq += v * v;
  }
#pragma unroll
  for (int m = 16; m >= 1; m >>= 1) {
    sum += __shfl_xor(sum, m, 32);
    sq  += __shfl_xor(sq,  m, 32);
  }
  int w = threadIdx.x >> 5;
  if ((threadIdx.x & 31) == 0) { red[w] = sum; red[8 + w] = sq; }
  __syncthreads();
  if (threadIdx.x == 0) {
    float s = 0.f, q = 0.f;
#pragma unroll
    for (int i = 0; i < 8; ++i) { s += red[i]; q += red[8 + i]; }
    float mu = s * (1.f / DD);
    red[16] = mu;
    red[17] = rsqrtf(q * (1.f / DD) - mu * mu + LN_EPS);
  }
  __syncthreads();
  const float mu = red[16], rstd = red[17];
#pragma unroll
  for (int i = 0; i < 2; ++i) {
    int c = threadIdx.x + i * 256;
    float y = (vals[i] - mu) * rstd * gamma[c] + beta[c];
    out32[(size_t)row * DD + c] = y;
    out16[(size_t)row * DD + c] = (_Float16)y;
  }
}

// ---------------------------------------------------------------------------
// Fused WMMA GEMM (+bias +optional residual) + LayerNorm (+optional head dot)
// wave tile 32x64, block tile 32x512. Ping-pong (X/Y) pipelined K loop:
// no fragment rotation copies, loads for step i+1 in flight during step i.
// K/32 must be even (K = 512 or 1024 here). launch_bounds keeps the whole
// working set (~200 VGPRs) in registers -- no spills.
// ---------------------------------------------------------------------------
__global__ void __launch_bounds__(256, 2)
gemm_ln(const _Float16* __restrict__ A,
        const _Float16* __restrict__ Wt,
        const float* __restrict__ bias,
        const float* __restrict__ resid,   // nullable
        const float* __restrict__ gamma,
        const float* __restrict__ beta,
        float* __restrict__ out32,         // nullable
        _Float16* __restrict__ out16,      // nullable
        const float* __restrict__ headw,   // nullable
        const float* __restrict__ headb,
        float* __restrict__ yout,
        int K) {
  __shared__ float s[32][DD];
  const int tid = threadIdx.x;
  const int wave = tid >> 5, lane = tid & 31, lr = lane & 15, hf = lane >> 4;
  const int rowbase = blockIdx.x * 32;
  const int colbase = wave * 64;
  const _Float16* At0 = A + (size_t)rowbase * K;
  const _Float16* At1 = A + (size_t)(rowbase + 16) * K;

  v8f acc[2][4];
#pragma unroll
  for (int t = 0; t < 2; ++t)
#pragma unroll
    for (int c = 0; c < 4; ++c) acc[t][c] = (v8f)(0.f);

  auto loadset = [&](int k0, v16h (&a)[2], v16h (&bf)[4]) {
    a[0] = load_frag(At0, K, k0);
    a[1] = load_frag(At1, K, k0);
#pragma unroll
    for (int c = 0; c < 4; ++c)
      bf[c] = load_frag(Wt + (size_t)(colbase + c * 16) * K, K, k0);
  };
  auto domm = [&](v16h (&a)[2], v16h (&bf)[4]) {
#pragma unroll
    for (int c = 0; c < 4; ++c) {
      acc[0][c] = wmma_f16(a[0], bf[c], acc[0][c]);
      acc[1][c] = wmma_f16(a[1], bf[c], acc[1][c]);
    }
  };

  v16h ax[2], bx[4], ay[2], by[4];
  loadset(0, ax, bx);
  for (int k0 = 0; k0 < K - 64; k0 += 64) {
    loadset(k0 + 32, ay, by);
    domm(ax, bx);
    loadset(k0 + 64, ax, bx);
    domm(ay, by);
  }
  loadset(K - 32, ay, by);
  domm(ax, bx);
  domm(ay, by);

  // C layout -> LDS row-major tile
#pragma unroll
  for (int t = 0; t < 2; ++t)
#pragma unroll
    for (int c = 0; c < 4; ++c)
#pragma unroll
      for (int r = 0; r < 8; ++r)
        s[t * 16 + hf * 8 + r][colbase + c * 16 + lr] = acc[t][c][r];
  __syncthreads();

  // 8 threads per row (32 rows x 256 threads); width-8 shuffles stay intra-wave
  const int row = tid >> 3, sub = tid & 7;
  const float* rres = resid ? resid + (size_t)(rowbase + row) * DD : (const float*)0;
  float sum = 0.f, sq = 0.f;
#pragma unroll 4
  for (int j = 0; j < 64; ++j) {
    int col = j * 8 + sub;
    float v = s[row][col] + bias[col] + (rres ? rres[col] : 0.f);
    s[row][col] = v;
    sum += v; sq += v * v;
  }
#pragma unroll
  for (int m = 4; m >= 1; m >>= 1) {
    sum += __shfl_xor(sum, m, 8);
    sq  += __shfl_xor(sq,  m, 8);
  }
  const float mu = sum * (1.f / DD);
  const float rstd = rsqrtf(sq * (1.f / DD) - mu * mu + LN_EPS);

  float ha = 0.f;
#pragma unroll 4
  for (int j = 0; j < 64; ++j) {
    int col = j * 8 + sub;
    float y = (s[row][col] - mu) * rstd * gamma[col] + beta[col];
    if (out32) out32[(size_t)(rowbase + row) * DD + col] = y;
    if (out16) out16[(size_t)(rowbase + row) * DD + col] = (_Float16)y;
    if (headw) ha += y * headw[col];
  }
  if (headw) {
#pragma unroll
    for (int m = 4; m >= 1; m >>= 1) ha += __shfl_xor(ha, m, 8);
    if (sub == 0) yout[rowbase + row] = ha + headb[0];
  }
}

// ---------------------------------------------------------------------------
// Generic WMMA GEMM + bias (+ReLU | +RoPE+scale | +transposed store for V)
// wave tile 32x64, block tile 32x512, grid.y covers N/512. Ping-pong K loop.
// ---------------------------------------------------------------------------
__global__ void __launch_bounds__(256, 2)
gemm_bias(const _Float16* __restrict__ A,
          const _Float16* __restrict__ Wt,
          const float* __restrict__ bias,
          _Float16* __restrict__ out,    // nullable
          _Float16* __restrict__ outT,   // nullable: [B,H,HD,seq]
          int N, int K, int ropeSeq, float scale, int relu,
          int seqN) {
  const int tid = threadIdx.x;
  const int wave = tid >> 5, lane = tid & 31, lr = lane & 15, hf = lane >> 4;
  const int rowbase = blockIdx.x * 32;
  const int colbase = blockIdx.y * 512 + wave * 64;
  const _Float16* At0 = A + (size_t)rowbase * K;
  const _Float16* At1 = A + (size_t)(rowbase + 16) * K;

  v8f acc[2][4];
#pragma unroll
  for (int t = 0; t < 2; ++t)
#pragma unroll
    for (int c = 0; c < 4; ++c) acc[t][c] = (v8f)(0.f);

  auto loadset = [&](int k0, v16h (&a)[2], v16h (&bf)[4]) {
    a[0] = load_frag(At0, K, k0);
    a[1] = load_frag(At1, K, k0);
#pragma unroll
    for (int c = 0; c < 4; ++c)
      bf[c] = load_frag(Wt + (size_t)(colbase + c * 16) * K, K, k0);
  };
  auto domm = [&](v16h (&a)[2], v16h (&bf)[4]) {
#pragma unroll
    for (int c = 0; c < 4; ++c) {
      acc[0][c] = wmma_f16(a[0], bf[c], acc[0][c]);
      acc[1][c] = wmma_f16(a[1], bf[c], acc[1][c]);
    }
  };

  v16h ax[2], bx[4], ay[2], by[4];
  loadset(0, ax, bx);
  for (int k0 = 0; k0 < K - 64; k0 += 64) {
    loadset(k0 + 32, ay, by);
    domm(ax, bx);
    loadset(k0 + 64, ax, bx);
    domm(ay, by);
  }
  loadset(K - 32, ay, by);
  domm(ax, bx);
  domm(ay, by);

#pragma unroll
  for (int c = 0; c < 4; ++c) {
    const int col = colbase + c * 16 + lr;
    const float bb = bias[col];
    const int dh = col & (HDIM - 1);
    const float invf = __expf(-(float)(dh & ~1) * ROPE_LN1E4_OVER_HD);
#pragma unroll
    for (int t = 0; t < 2; ++t) {
#pragma unroll
      for (int r = 0; r < 8; ++r) {
        const int row = rowbase + t * 16 + hf * 8 + r;
        float v = acc[t][c][r] + bb;
        if (ropeSeq) {
          const int pos = row % ropeSeq;   // row = b*seq + pos
          float sn, cs;
          __sincosf((float)pos * invf, &sn, &cs);
          const float other = __shfl_xor(v, 1, 32);  // even<->odd column pair
          v = (dh & 1) ? (other * sn + v * cs) : (v * cs - other * sn);
        }
        v *= scale;
        if (relu) v = fmaxf(v, 0.f);
        const _Float16 hv = (_Float16)v;
        if (out) out[(size_t)row * N + col] = hv;
        if (outT) {
          const int bi = row / seqN, n = row - bi * seqN;
          const int h = col >> 6, d = col & (HDIM - 1);
          outT[((size_t)(bi * HH + h) * HDIM + d) * seqN + n] = hv;
        }
      }
    }
  }
}

// ---------------------------------------------------------------------------
// Flash attention: grid (G/128, B*H), block 128 (4 waves x 32 query rows).
// Key fragments ping-pong across iterations (no copies); V fragments for the
// current block issue at the top of each step and are consumed after the
// VALU-heavy softmax, which hides their latency.
// ---------------------------------------------------------------------------
__global__ void __launch_bounds__(128, 2)
attn_flash(const _Float16* __restrict__ q,
           const _Float16* __restrict__ k,
           const _Float16* __restrict__ vT,
           _Float16* __restrict__ out) {
  __shared__ __align__(16) _Float16 pbuf[4][32 * 32];
  const int tid = threadIdx.x;
  const int wave = tid >> 5, lane = tid & 31, lr = lane & 15, hf = lane >> 4;
  const int bh = blockIdx.y, b = bh >> 3, h = bh & 7;
  const int g0 = blockIdx.x * 128 + wave * 32;

  const _Float16* kbase = k + (size_t)b * NN * DD + h * HDIM;
  const _Float16* vbase = vT + (size_t)bh * HDIM * NN;
  _Float16* pw = pbuf[wave];

  v16h aq[2][2];
#pragma unroll
  for (int t = 0; t < 2; ++t) {
    const _Float16* qtile = q + ((size_t)b * GG + g0 + t * 16) * DD + h * HDIM;
    aq[t][0] = load_frag(qtile, DD, 0);
    aq[t][1] = load_frag(qtile, DD, 32);
  }

  float m[2][8], l[2][8];
  v8f o[2][4];
#pragma unroll
  for (int t = 0; t < 2; ++t) {
#pragma unroll
    for (int r = 0; r < 8; ++r) { m[t][r] = -1e30f; l[t][r] = 0.f; }
#pragma unroll
    for (int c = 0; c < 4; ++c) o[t][c] = (v8f)(0.f);
  }

  auto loadK = [&](int nb, v16h (&kf)[4]) {
    kf[0] = load_frag(kbase + (size_t)nb * DD, DD, 0);
    kf[1] = load_frag(kbase + (size_t)nb * DD, DD, 32);
    kf[2] = load_frag(kbase + (size_t)(nb + 16) * DD, DD, 0);
    kf[3] = load_frag(kbase + (size_t)(nb + 16) * DD, DD, 32);
  };

  // One flash step for key block nb using fragments kc; optionally issues the
  // loads for the next block into kn right after the QK^T WMMAs.
  auto step = [&](int nb, v16h (&kc)[4], v16h (&kn)[4], int nbn, bool ld) {
    // V fragments for this block: independent, issue early
    v16h vf[4];
#pragma unroll
    for (int c = 0; c < 4; ++c)
      vf[c] = load_frag(vbase + (size_t)(c * 16) * NN, NN, nb);

    v8f sc[2][2];
#pragma unroll
    for (int t = 0; t < 2; ++t) {
      sc[t][0] = wmma_f16(aq[t][0], kc[0], (v8f)(0.f));
      sc[t][0] = wmma_f16(aq[t][1], kc[1], sc[t][0]);
      sc[t][1] = wmma_f16(aq[t][0], kc[2], (v8f)(0.f));
      sc[t][1] = wmma_f16(aq[t][1], kc[3], sc[t][1]);
    }
    if (ld) loadK(nbn, kn);

    // online softmax (rows live across the 16 lanes of each half-wave)
#pragma unroll
    for (int t = 0; t < 2; ++t) {
      float alpha[8];
#pragma unroll
      for (int r = 0; r < 8; ++r) {
        float mx = fmaxf(sc[t][0][r], sc[t][1][r]);
#pragma unroll
        for (int msk = 8; msk >= 1; msk >>= 1)
          mx = fmaxf(mx, __shfl_xor(mx, msk, 16));
        const float mn = fmaxf(m[t][r], mx);
        alpha[r] = __expf(m[t][r] - mn);
        m[t][r] = mn;
        const float p0 = __expf(sc[t][0][r] - mn);
        const float p1 = __expf(sc[t][1][r] - mn);
        float rs = p0 + p1;
#pragma unroll
        for (int msk = 8; msk >= 1; msk >>= 1) rs += __shfl_xor(rs, msk, 16);
        l[t][r] = l[t][r] * alpha[r] + rs;
        const int rr = t * 16 + hf * 8 + r;
        pw[rr * 32 + lr]      = (_Float16)p0;  // C layout -> row-major P tile
        pw[rr * 32 + 16 + lr] = (_Float16)p1;
      }
#pragma unroll
      for (int c = 0; c < 4; ++c)
#pragma unroll
        for (int r = 0; r < 8; ++r) o[t][c][r] *= alpha[r];
    }

    // O += P[32x32] @ V[32x64]; P reloaded from LDS in A layout (ds_load_b128)
    const v16h ap0 = load_frag(pw, 32, 0);
    const v16h ap1 = load_frag(pw + 16 * 32, 32, 0);
#pragma unroll
    for (int c = 0; c < 4; ++c) {
      o[0][c] = wmma_f16(ap0, vf[c], o[0][c]);
      o[1][c] = wmma_f16(ap1, vf[c], o[1][c]);
    }
  };

  v16h kx[4], ky[4];
  loadK(0, kx);
  for (int nb = 0; nb < NN - 64; nb += 64) {
    step(nb,      kx, ky, nb + 32, true);
    step(nb + 32, ky, kx, nb + 64, true);
  }
  step(NN - 64, kx, ky, NN - 32, true);
  step(NN - 32, ky, kx, 0, false);

#pragma unroll
  for (int t = 0; t < 2; ++t)
#pragma unroll
    for (int c = 0; c < 4; ++c) {
      const int col = h * HDIM + c * 16 + lr;
#pragma unroll
      for (int r = 0; r < 8; ++r) {
        const int row = g0 + t * 16 + hf * 8 + r;
        out[((size_t)b * GG + row) * DD + col] = (_Float16)(o[t][c][r] / l[t][r]);
      }
    }
}

// ---------------------------------------------------------------------------
// Host-side orchestration
// ---------------------------------------------------------------------------
extern "C" void kernel_launch(void* const* d_in, const int* in_sizes, int n_in,
                              void* d_out, int out_size, void* d_ws, size_t ws_size,
                              hipStream_t stream) {
  (void)in_sizes; (void)n_in; (void)out_size; (void)ws_size;
  const float* img_feats   = (const float*)d_in[0];
  const float* gene_values = (const float*)d_in[1];
  const float* img_proj_w  = (const float*)d_in[2];
  const float* img_proj_b  = (const float*)d_in[3];
  const float* img_norm_g  = (const float*)d_in[4];
  const float* img_norm_b  = (const float*)d_in[5];
  const float* gene_w      = (const float*)d_in[6];
  const float* gene_b      = (const float*)d_in[7];
  const float* gene_norm_g = (const float*)d_in[8];
  const float* gene_norm_b = (const float*)d_in[9];
  const float* wq          = (const float*)d_in[10];
  const float* bq          = (const float*)d_in[11];
  const float* wk          = (const float*)d_in[12];
  const float* bk          = (const float*)d_in[13];
  const float* wv          = (const float*)d_in[14];
  const float* bv          = (const float*)d_in[15];
  const float* wo          = (const float*)d_in[16];
  const float* bo          = (const float*)d_in[17];
  const float* ln1_g       = (const float*)d_in[18];
  const float* ln1_b       = (const float*)d_in[19];
  const float* ff1_w       = (const float*)d_in[20];
  const float* ff1_b       = (const float*)d_in[21];
  const float* ff2_w       = (const float*)d_in[22];
  const float* ff2_b       = (const float*)d_in[23];
  const float* ln2_g       = (const float*)d_in[24];
  const float* ln2_b       = (const float*)d_in[25];
  const float* head_w      = (const float*)d_in[26];
  const float* head_b      = (const float*)d_in[27];

  float* y_out = (float*)d_out;          // [B*G]
  float* x_out = (float*)d_out + MG;     // [B*G, D]

  char* ws = (char*)d_ws;
  size_t off = 0;
  auto alloc = [&](size_t bytes) { char* p = ws + off; off += (bytes + 255) & ~(size_t)255; return p; };

  _Float16* Aimg   = (_Float16*)alloc((size_t)MI * CIN * 2);   // img_feats f16
  _Float16* WtImg  = (_Float16*)alloc((size_t)DD * CIN * 2);   // [512][1024]
  _Float16* Wtq    = (_Float16*)alloc((size_t)DD * DD * 2);
  _Float16* Wtk    = (_Float16*)alloc((size_t)DD * DD * 2);
  _Float16* Wtv    = (_Float16*)alloc((size_t)DD * DD * 2);
  _Float16* Wto    = (_Float16*)alloc((size_t)DD * DD * 2);
  _Float16* Wtff1  = (_Float16*)alloc((size_t)DFF * DD * 2);   // [1024][512]
  _Float16* Wtff2  = (_Float16*)alloc((size_t)DD * DFF * 2);   // [512][1024]
  float*    gene32 = (float*)   alloc((size_t)MG * DD * 4);
  _Float16* gene16 = (_Float16*)alloc((size_t)MG * DD * 2);
  _Float16* img16  = (_Float16*)alloc((size_t)MI * DD * 2);
  _Float16* qbuf   = (_Float16*)alloc((size_t)MG * DD * 2);
  _Float16* kbuf   = (_Float16*)alloc((size_t)MI * DD * 2);
  _Float16* vTbuf  = (_Float16*)alloc((size_t)MI * DD * 2);    // [B,H,HD,N]
  _Float16* attn16 = (_Float16*)alloc((size_t)MG * DD * 2);
  float*    x1_32  = (float*)   alloc((size_t)MG * DD * 4);
  _Float16* x1_16  = (_Float16*)alloc((size_t)MG * DD * 2);
  _Float16* h16    = (_Float16*)alloc((size_t)MG * DFF * 2);

  // 1) activations -> f16
  {
    int n = MI * CIN;
    cvt_f32_to_f16<<<(n + 255) / 256, 256, 0, stream>>>(img_feats, Aimg, n);
  }
  // 2) weights -> f16, transposed to [N][K]
  cvt_transpose_w<<<(CIN * DD + 255) / 256, 256, 0, stream>>>(img_proj_w, WtImg, CIN, DD);
  cvt_transpose_w<<<(DD * DD + 255) / 256, 256, 0, stream>>>(wq, Wtq, DD, DD);
  cvt_transpose_w<<<(DD * DD + 255) / 256, 256, 0, stream>>>(wk, Wtk, DD, DD);
  cvt_transpose_w<<<(DD * DD + 255) / 256, 256, 0, stream>>>(wv, Wtv, DD, DD);
  cvt_transpose_w<<<(DD * DD + 255) / 256, 256, 0, stream>>>(wo, Wto, DD, DD);
  cvt_transpose_w<<<(DD * DFF + 255) / 256, 256, 0, stream>>>(ff1_w, Wtff1, DD, DFF);
  cvt_transpose_w<<<(DFF * DD + 255) / 256, 256, 0, stream>>>(ff2_w, Wtff2, DFF, DD);

  // 3) gene embedding + LN
  gene_embed_ln<<<MG, 256, 0, stream>>>(gene_values, gene_w, gene_b,
                                        gene_norm_g, gene_norm_b, gene32, gene16);

  // 4) img projection + LN  (M=16384, K=1024)
  gemm_ln<<<MI / 32, 256, 0, stream>>>(Aimg, WtImg, img_proj_b, (const float*)0,
                                       img_norm_g, img_norm_b,
                                       (float*)0, img16,
                                       (const float*)0, (const float*)0, (float*)0,
                                       CIN);

  // 5) Q (RoPE over G, pre-scaled by 1/sqrt(HD)), K (RoPE over N), V (transposed)
  gemm_bias<<<dim3(MG / 32, 1), 256, 0, stream>>>(gene16, Wtq, bq, qbuf, (_Float16*)0,
                                                  DD, DD, GG, 0.125f, 0, GG);
  gemm_bias<<<dim3(MI / 32, 1), 256, 0, stream>>>(img16, Wtk, bk, kbuf, (_Float16*)0,
                                                  DD, DD, NN, 1.0f, 0, NN);
  gemm_bias<<<dim3(MI / 32, 1), 256, 0, stream>>>(img16, Wtv, bv, (_Float16*)0, vTbuf,
                                                  DD, DD, 0, 1.0f, 0, NN);

  // 6) flash attention: grid (G/128, B*H), 4 waves x 32 queries per block
  attn_flash<<<dim3(GG / 128, BB * HH), 128, 0, stream>>>(qbuf, kbuf, vTbuf, attn16);

  // 7) WO + residual(gene) + LN1
  gemm_ln<<<MG / 32, 256, 0, stream>>>(attn16, Wto, bo, gene32,
                                       ln1_g, ln1_b, x1_32, x1_16,
                                       (const float*)0, (const float*)0, (float*)0,
                                       DD);

  // 8) FF1 + ReLU  (M=32768, K=512, N=1024)
  gemm_bias<<<dim3(MG / 32, DFF / 512), 256, 0, stream>>>(x1_16, Wtff1, ff1_b, h16,
                                                          (_Float16*)0, DFF, DD,
                                                          0, 1.0f, 1, GG);

  // 9) FF2 + residual(x1) + LN2 + head projection -> outputs
  gemm_ln<<<MG / 32, 256, 0, stream>>>(h16, Wtff2, ff2_b, x1_32,
                                       ln2_g, ln2_b, x_out, (_Float16*)0,
                                       head_w, head_b, y_out,
                                       DFF);
}
